// LSTM_58695023068072
// MI455X (gfx1250) — matrix operand order repaired and checked by reading, compiled, not run on hardware
//
#include <hip/hip_runtime.h>
#include <hip/hip_bf16.h>
#include <hip/hip_fp16.h>

typedef __attribute__((ext_vector_type(16))) _Float16 v16h;
typedef __attribute__((ext_vector_type(8)))  float    v8f;

// TDM descriptor vector types (per CDNA5_HIP.md probe results)
typedef __attribute__((ext_vector_type(4))) unsigned tdm_u32x4;
typedef __attribute__((ext_vector_type(8))) int      tdm_i32x8;
typedef __attribute__((ext_vector_type(4))) int      tdm_i32x4;

#if __has_include(<hip/amd_detail/amd_gfx1250_TDM.h>)
#define TDM_SIX_ARGS 1
#else
#define TDM_SIX_ARGS 0
#endif

#define BATCH 256
#define TSTEPS 512
#define HDIM 1024
#define NGATE 4
#define NWG 64            // HDIM / 16 column-owner workgroups
#define THREADS 256
#define CPW 16            // h-columns per WG
// workspace layout (bytes)
#define PK_BYTES   (NGATE * HDIM * HDIM * 2)          // 8 MB packed f16 weights
#define HBUF_HALF  (BATCH * HDIM)                     // one h buffer, halves
#define HBUF_BYTES (HBUF_HALF * 2)
#define WS_PK    0
#define WS_H0    (PK_BYTES)
#define WS_H1    (PK_BYTES + HBUF_BYTES)
#define WS_SYNC  (PK_BYTES + 2 * HBUF_BYTES)
#define WS_XT    (WS_SYNC + 256)                      // x transposed [T][B] fp32

__device__ __forceinline__ float sigmoid_fast(float v) {
    return 1.0f / (1.0f + __expf(-v));
}
__device__ __forceinline__ float tanh_fast(float v) {
    float e = __expf(2.0f * v);
    return (e - 1.0f) / (e + 1.0f);
}

// ---------------- init: zero h double-buffer and sync words ----------------
__global__ void lstm_init_kernel(_Float16* h0, _Float16* h1, unsigned* sync) {
    unsigned idx = blockIdx.x * blockDim.x + threadIdx.x;
    if (idx < HBUF_HALF) { h0[idx] = (_Float16)0.0f; h1[idx] = (_Float16)0.0f; }
    if (idx < 16) sync[idx] = 0u;
}

// ---------------- x transpose: [B][T] -> [T][B] for coalesced step loads ---
__global__ void x_transpose_kernel(const float* __restrict__ x, float* __restrict__ xT) {
    int t = blockIdx.x;       // 0..511
    int b = threadIdx.x;      // 0..255
    xT[t * BATCH + b] = x[b * TSTEPS + t];
}

// ---------------- pack: Wh fp32 -> f16, per-WG column-sliced layout --------
// pk[((wg*4 + gate)*1024 + k)*16 + c] = Wh_gate[k][wg*16 + c]
__global__ void lstm_pack_kernel(const float* __restrict__ wgh,
                                 const float* __restrict__ wih,
                                 const float* __restrict__ wfh,
                                 const float* __restrict__ woh,
                                 _Float16* __restrict__ pk) {
    unsigned idx = blockIdx.x * blockDim.x + threadIdx.x;  // < 4*1024*1024
    unsigned c    = idx & 15u;
    unsigned k    = (idx >> 4) & 1023u;
    unsigned gate = (idx >> 14) & 3u;
    unsigned wg   = idx >> 16;
    const float* src = (gate == 0) ? wgh : (gate == 1) ? wih : (gate == 2) ? wfh : woh;
    pk[idx] = (_Float16)src[k * HDIM + wg * CPW + c];
}

// ---------------- persistent LSTM kernel -----------------------------------
__global__ void __launch_bounds__(THREADS)
lstm_persistent_kernel(const float* __restrict__ xT,    // [T][B]
                       const float* __restrict__ wgx, const float* __restrict__ wix,
                       const float* __restrict__ wfx, const float* __restrict__ wox,
                       const float* __restrict__ bg,  const float* __restrict__ bi,
                       const float* __restrict__ bf,  const float* __restrict__ bo,
                       const _Float16* __restrict__ pk,
                       _Float16* __restrict__ hbuf0,   // t-even read buffer
                       _Float16* __restrict__ hbuf1,   // t-odd  read buffer
                       unsigned* __restrict__ sync) {
    extern __shared__ _Float16 wlds[];                 // [4][1024][16] = 128 KB
    __shared__ float xs[BATCH];                        // x_t column
    __shared__ float wxs[NGATE * CPW];                 // Wx row slice (step-invariant)
    __shared__ float bsh[NGATE * CPW];                 // bias slice   (step-invariant)

    const int tid  = threadIdx.x;
    const int wg   = blockIdx.x;                       // owns h-cols [wg*16, wg*16+16)
    const int wave = tid >> 5;                         // 0..7 (wave32)
    const int lane = tid & 31;
    const int m_base = wave * 32;                      // 32 batch rows per wave
    const int col  = lane & 15;                        // h-col within the 16-wide tile
    const int rsel = (lane >> 4) * 8;                  // C/D layout row selector

    // ---- stage step-invariant Wx row + bias slices ----
    if (tid < NGATE * CPW) {
        int gate = tid >> 4, c = tid & 15;
        int j = wg * CPW + c;
        const float* wx = (gate == 0) ? wgx : (gate == 1) ? wix : (gate == 2) ? wfx : wox;
        const float* bb = (gate == 0) ? bg  : (gate == 1) ? bi  : (gate == 2) ? bf  : bo;
        wxs[tid] = wx[j];
        bsh[tid] = bb[j];
    }

    // ---- load this WG's Wh column slice into LDS (resident for 512 steps) ----
#if __has_builtin(__builtin_amdgcn_tensor_load_to_lds) && __has_builtin(__builtin_amdgcn_s_wait_tensorcnt)
    if (tid == 0) {
        // Tensor Data Mover: one 128KB contiguous tile (8192 dwords x 4 rows),
        // global pk + wg*128KB  ->  LDS wlds. D# per CDNA5 ISA.
        unsigned long long gaddr =
            (unsigned long long)(uintptr_t)pk + (unsigned long long)wg * (NGATE * HDIM * CPW * 2);
        unsigned ldsOff = (unsigned)(uintptr_t)(void*)wlds;   // low 32 bits = LDS byte offset
        tdm_u32x4 g0;
        g0[0] = 0x1u;                                         // count=1, user mode
        g0[1] = ldsOff;                                       // lds_addr
        g0[2] = (unsigned)(gaddr & 0xFFFFFFFFull);            // global_addr[31:0]
        g0[3] = (unsigned)((gaddr >> 32) & 0x01FFFFFFull)     // global_addr[56:32]
              | 0x80000000u;                                  // type=2 ("image")
        tdm_i32x8 g1;
        g1[0] = 0x00020000;                  // wg_mask=0, data_size=2 (4-byte units)
        g1[1] = (int)(0x2000u << 16);        // tensor_dim0 lo16 = 8192
        g1[2] = (int)(4u << 16);             // tensor_dim0 hi=0 | tensor_dim1 lo16 = 4
        g1[3] = (int)(0x2000u << 16);        // tensor_dim1 hi=0 | tile_dim0 = 8192
        g1[4] = 4;                           // tile_dim1 = 4, tile_dim2 = 0
        g1[5] = 8192;                        // tensor_dim0_stride[31:0]
        g1[6] = 0;                           // stride hi | tensor_dim1_stride lo
        g1[7] = 0;
        tdm_i32x4 gz4 = {0, 0, 0, 0};
#if TDM_SIX_ARGS
        tdm_i32x8 gz8 = {0, 0, 0, 0, 0, 0, 0, 0};
        __builtin_amdgcn_tensor_load_to_lds(g0, g1, gz4, gz4, gz8, 0);
#else
        __builtin_amdgcn_tensor_load_to_lds(g0, g1, gz4, gz4, 0);
#endif
        __builtin_amdgcn_s_wait_tensorcnt(0);
    }
#else
    {
        const v16h* pv = (const v16h*)pk + (size_t)wg * (NGATE * HDIM * CPW / 16);
        v16h* wv = (v16h*)wlds;
        for (int i = tid; i < NGATE * HDIM * CPW / 16; i += THREADS) wv[i] = pv[i];
    }
#endif
    __syncthreads();

    const v8f zerov = {0.f, 0.f, 0.f, 0.f, 0.f, 0.f, 0.f, 0.f};
    v8f creg[2]; creg[0] = zerov; creg[1] = zerov;     // cell state lives in VGPRs

    for (int t = 0; t < TSTEPS; ++t) {
        const _Float16* hread  = (t & 1) ? hbuf1 : hbuf0;
        _Float16*       hwrite = (t & 1) ? hbuf0 : hbuf1;

        // stage x_t column into LDS (coalesced from transposed x)
        xs[tid] = xT[t * BATCH + tid];
        __syncthreads();

        v8f acc[NGATE][2];
        #pragma unroll
        for (int g = 0; g < NGATE; ++g) { acc[g][0] = zerov; acc[g][1] = zerov; }

        // ---- z_tile = h @ Wh_slice via WMMA, B matrices from LDS ----
        const int r0 = m_base + (lane & 15);
        const int ksel = (lane >> 4) * 16;
        for (int k0 = 0; k0 < HDIM; k0 += 32) {
            v16h a0 = *(const v16h*)(hread + (size_t)r0 * HDIM + k0 + ksel);
            v16h a1 = *(const v16h*)(hread + (size_t)(r0 + 16) * HDIM + k0 + ksel);
            #pragma unroll
            for (int g = 0; g < NGATE; ++g) {
                v16h bfr;
                #pragma unroll
                for (int s = 0; s < 16; ++s)
                    bfr[s] = wlds[((size_t)g * HDIM + k0 + ksel + s) * CPW + col];
                acc[g][0] = __builtin_amdgcn_wmma_f32_16x16x32_f16(
                    false, a0, false, bfr, (short)0, acc[g][0], false, false);
                acc[g][1] = __builtin_amdgcn_wmma_f32_16x16x32_f16(
                    false, a1, false, bfr, (short)0, acc[g][1], false, false);
            }
        }

        // ---- gate activations + cell update (c stays in registers) ----
        #pragma unroll
        for (int tile = 0; tile < 2; ++tile) {
            #pragma unroll
            for (int v = 0; v < 8; ++v) {
                int row = m_base + tile * 16 + rsel + v;
                float xv = xs[row];
                float zg = acc[0][tile][v] + xv * wxs[ 0 + col] + bsh[ 0 + col];
                float zi = acc[1][tile][v] + xv * wxs[16 + col] + bsh[16 + col];
                float zf = acc[2][tile][v] + xv * wxs[32 + col] + bsh[32 + col];
                float zo = acc[3][tile][v] + xv * wxs[48 + col] + bsh[48 + col];
                float gv = tanh_fast(zg);
                float iv = sigmoid_fast(zi);
                float fv = sigmoid_fast(zf);
                float ov = sigmoid_fast(zo);
                float cv = gv * iv + creg[tile][v] * fv;
                creg[tile][v] = cv;
                float hv = tanh_fast(cv) * ov;
                hwrite[(size_t)row * HDIM + wg * CPW + col] = (_Float16)hv;
            }
        }

        // ---- grid-wide generation barrier (all 64 WGs resident) ----
        __threadfence();
        __syncthreads();
        if (tid == 0) {
            unsigned gen = __hip_atomic_load(&sync[1], __ATOMIC_ACQUIRE, __HIP_MEMORY_SCOPE_AGENT);
            unsigned arr = __hip_atomic_fetch_add(&sync[0], 1u, __ATOMIC_ACQ_REL, __HIP_MEMORY_SCOPE_AGENT);
            if (arr == NWG - 1) {
                __hip_atomic_store(&sync[0], 0u, __ATOMIC_RELAXED, __HIP_MEMORY_SCOPE_AGENT);
                __hip_atomic_fetch_add(&sync[1], 1u, __ATOMIC_ACQ_REL, __HIP_MEMORY_SCOPE_AGENT);
            } else {
                while (__hip_atomic_load(&sync[1], __ATOMIC_ACQUIRE, __HIP_MEMORY_SCOPE_AGENT) == gen) {
                    __builtin_amdgcn_s_sleep(2);
                }
            }
        }
        __syncthreads();
    }
}

// ---------------- projection + softmax: out = softmax(h_T @ Wp + bp) -------
__global__ void lstm_proj_kernel(const _Float16* __restrict__ hT,  // buffer 0 (T even)
                                 const float* __restrict__ wph,    // [H][10]
                                 const float* __restrict__ bp,     // [10]
                                 float* __restrict__ out) {        // [B][10]
    const int b = blockIdx.x, tid = threadIdx.x;
    float acc[10];
    #pragma unroll
    for (int c = 0; c < 10; ++c) acc[c] = 0.0f;
    for (int h = tid; h < HDIM; h += THREADS) {
        float hv = (float)hT[(size_t)b * HDIM + h];
        #pragma unroll
        for (int c = 0; c < 10; ++c) acc[c] += hv * wph[h * 10 + c];
    }
    __shared__ float red[THREADS];
    __shared__ float logits[10];
    for (int c = 0; c < 10; ++c) {
        red[tid] = acc[c];
        __syncthreads();
        for (int s = THREADS / 2; s > 0; s >>= 1) {
            if (tid < s) red[tid] += red[tid + s];
            __syncthreads();
        }
        if (tid == 0) logits[c] = red[0] + bp[c];
        __syncthreads();
    }
    if (tid == 0) {
        float m = logits[0];
        #pragma unroll
        for (int c = 1; c < 10; ++c) m = fmaxf(m, logits[c]);
        float s = 0.0f, e[10];
        #pragma unroll
        for (int c = 0; c < 10; ++c) { e[c] = __expf(logits[c] - m); s += e[c]; }
        float inv = 1.0f / s;
        #pragma unroll
        for (int c = 0; c < 10; ++c) out[b * 10 + c] = e[c] * inv;
    }
}

extern "C" void kernel_launch(void* const* d_in, const int* in_sizes, int n_in,
                              void* d_out, int out_size, void* d_ws, size_t ws_size,
                              hipStream_t stream) {
    const float* x   = (const float*)d_in[0];
    const float* wgx = (const float*)d_in[1];
    const float* wgh = (const float*)d_in[2];
    const float* bg  = (const float*)d_in[3];
    const float* wix = (const float*)d_in[4];
    const float* wih = (const float*)d_in[5];
    const float* bi  = (const float*)d_in[6];
    const float* wfx = (const float*)d_in[7];
    const float* wfh = (const float*)d_in[8];
    const float* bf  = (const float*)d_in[9];
    const float* wox = (const float*)d_in[10];
    const float* woh = (const float*)d_in[11];
    const float* bo  = (const float*)d_in[12];
    const float* wph = (const float*)d_in[13];
    const float* bp  = (const float*)d_in[14];

    char* ws = (char*)d_ws;
    _Float16* pk    = (_Float16*)(ws + WS_PK);
    _Float16* h0    = (_Float16*)(ws + WS_H0);
    _Float16* h1    = (_Float16*)(ws + WS_H1);
    unsigned* syncw = (unsigned*)(ws + WS_SYNC);
    float*    xTb   = (float*)(ws + WS_XT);

    // 1) zero h double-buffer + sync words
    lstm_init_kernel<<<(HBUF_HALF + THREADS - 1) / THREADS, THREADS, 0, stream>>>(h0, h1, syncw);

    // 2a) transpose x to [T][B] for coalesced per-step staging
    x_transpose_kernel<<<TSTEPS, BATCH, 0, stream>>>(x, xTb);

    // 2b) pack Wh fp32 -> f16 column-sliced layout (8 MB)
    lstm_pack_kernel<<<(NGATE * HDIM * HDIM) / THREADS, THREADS, 0, stream>>>(wgh, wih, wfh, woh, pk);

    // 3) persistent recurrent kernel: 64 WGs, 128 KB dynamic LDS each
    lstm_persistent_kernel<<<NWG, THREADS, NGATE * HDIM * CPW * sizeof(_Float16), stream>>>(
        xTb, wgx, wix, wfx, wox, bg, bi, bf, bo, pk, h0, h1, syncw);

    // 4) projection + softmax (TSTEPS even -> final h in buffer 0)
    lstm_proj_kernel<<<BATCH, THREADS, 0, stream>>>(h0, wph, bp, (float*)d_out);
}
